// CA_GAT_enriched_shadow_nums_1056561955501
// MI455X (gfx1250) — compile-verified, matrix operand order checked by reading, not run on hardware
//
#include <hip/hip_runtime.h>
#include <cstdint>
#include <cstddef>

// ---------- types ----------
typedef __attribute__((ext_vector_type(16))) __bf16 v16bf;
typedef __attribute__((ext_vector_type(8)))  float  v8f;
typedef __attribute__((ext_vector_type(4))) unsigned int v4u;
typedef __attribute__((ext_vector_type(4))) int v4i;
typedef __attribute__((ext_vector_type(8))) int v8i;

#if __has_include(<hip/amd_detail/amd_gfx1250_TDM.h>)
#define TDM_6ARG 1
#else
#define TDM_6ARG 0
#endif

// fp32 -> bf16 (round to nearest even)
static __device__ __forceinline__ unsigned short f2bf(float f) {
    unsigned u = __float_as_uint(f);
    unsigned r = ((u >> 16) & 1u) + 0x7FFFu;
    return (unsigned short)((u + r) >> 16);
}

// order-preserving float<->uint for atomicMax on floats (incl. negatives / -inf)
static __device__ __forceinline__ unsigned encf(float f) {
    int b = __float_as_int(f);
    return (b >= 0) ? ((unsigned)b | 0x80000000u) : ~(unsigned)b;
}
static __device__ __forceinline__ float decf(unsigned u) {
    unsigned b = (u & 0x80000000u) ? (u ^ 0x80000000u) : ~u;
    return __uint_as_float(b);
}

static __device__ __forceinline__ float wred(float v) {
    #pragma unroll
    for (int off = 16; off > 0; off >>= 1) v += __shfl_down(v, off, 32);
    return v;
}

// ---------- TDM: DMA one [128 rows x 32 bf16] tile (row stride ld_elems) into LDS ----------
// pad_enable: +4 DWORDs every 16 DWORDs => LDS row stride = 64B + 16B = 80B (40 ushorts),
// matching the 16B-aligned ds_load_b128 fragment layout and avoiding bank conflicts.
static __device__ __forceinline__ void tdm_load_tile(unsigned lds_off,
                                                     const unsigned short* gptr,
                                                     unsigned rows_valid,
                                                     unsigned ld_elems) {
    unsigned long long ga = (unsigned long long)(uintptr_t)gptr;
    v4u g0;
    g0[0] = 1u;                                        // count=1 (valid user descriptor)
    g0[1] = lds_off;                                   // lds_addr (bytes)
    g0[2] = (unsigned)(ga & 0xFFFFFFFFu);              // global_addr[31:0]
    g0[3] = (unsigned)((ga >> 32) & 0x01FFFFFFu) | (2u << 30);  // addr[56:32] | type=2
    v8i g1;
    g1[0] = (int)((1u << 16)      // data_size = 1 -> 2 bytes
                | (1u << 20)      // pad_enable
                | (3u << 22)      // pad_interval: 16 DWORDs (one 32-bf16 row)
                | (3u << 25));    // pad_amount: 4 DWORDs
    g1[1] = (int)(32u << 16);                                     // tensor_dim0 = 32
    g1[2] = (int)((rows_valid & 0xFFFFu) << 16);                  // tensor_dim1 lo16
    g1[3] = (int)(((rows_valid >> 16) & 0xFFFFu) | (32u << 16));  // dim1 hi16 | tile_dim0=32
    g1[4] = (int)128u;                                            // tile_dim1 = 128 rows
    g1[5] = (int)ld_elems;                                        // tensor_dim0_stride lo32
    g1[6] = 0;
    g1[7] = 0;
    v4i z4 = {0, 0, 0, 0};
#if TDM_6ARG
    v8i z8 = {0, 0, 0, 0, 0, 0, 0, 0};
    __builtin_amdgcn_tensor_load_to_lds(g0, g1, z4, z4, z8, 0);
#else
    __builtin_amdgcn_tensor_load_to_lds(g0, g1, z4, z4, 0);
#endif
}

// ---------- one-time f32 -> bf16 conversion of activation matrices ----------
__global__ void cvt_bf16_kernel(const float* __restrict__ src,
                                unsigned short* __restrict__ dst, long long n8) {
    long long i = (long long)blockIdx.x * 256 + threadIdx.x;
    if (i >= n8) return;
    const float4* p = (const float4*)(src + i * 8);
    float4 a = p[0], b = p[1];
    uint4 o;
    o.x = (unsigned)f2bf(a.x) | ((unsigned)f2bf(a.y) << 16);
    o.y = (unsigned)f2bf(a.z) | ((unsigned)f2bf(a.w) << 16);
    o.z = (unsigned)f2bf(b.x) | ((unsigned)f2bf(b.y) << 16);
    o.w = (unsigned)f2bf(b.z) | ((unsigned)f2bf(b.w) << 16);
    *(uint4*)(dst + i * 8) = o;
}

// ---------- small prep: W_comb = W_num @ W4, b_comb = b_num @ W4 ----------
__global__ void wcomb_kernel(const float* __restrict__ Wnum, const float* __restrict__ bnum,
                             const float* __restrict__ W4, float* __restrict__ Wc,
                             float* __restrict__ bc) {
    int idx = blockIdx.x * 256 + threadIdx.x;
    if (idx < 192 * 78) {
        int k = idx / 78, c = idx % 78;
        float acc = 0.f;
        for (int j = 0; j < 768; ++j) acc += Wnum[k * 768 + j] * W4[j * 78 + c];
        Wc[idx] = acc;
    } else if (idx < 192 * 78 + 78) {
        int c = idx - 192 * 78;
        float acc = 0.f;
        for (int j = 0; j < 768; ++j) acc += bnum[j] * W4[j * 78 + c];
        bc[c] = acc;
    }
}

// ---------- pack W (f32 [nmat][K][78]) into per-lane WMMA B-fragment bf16 layout ----------
// Bp layout: [(m*5 + ntile)*ktiles + kt][lane(32)][i(16)] bf16
__global__ void pack_w_kernel(const float* __restrict__ W, unsigned short* __restrict__ Bp,
                              int nmat, int ktiles, int Kdim, int Cdim) {
    int idx = blockIdx.x * 256 + threadIdx.x;
    int total = nmat * 5 * ktiles * 32;
    if (idx >= total) return;
    int lane = idx & 31;
    int t = idx >> 5;
    int kt = t % ktiles; t /= ktiles;
    int ntile = t % 5;   int m = t / 5;
    int nn = ntile * 16 + (lane & 15);
    int half = lane >> 4;
    unsigned short* dst = Bp + (size_t)idx * 16;
    #pragma unroll
    for (int i = 0; i < 16; ++i) {
        int kk = (i < 8) ? (half * 8 + i) : (16 + half * 8 + (i - 8));
        int k = kt * 32 + kk;
        float v = 0.f;
        if (nn < Cdim && k < Kdim) v = W[((size_t)m * Kdim + k) * Cdim + nn];
        dst[i] = f2bf(v);
    }
}

// ---------- WMMA GEMM: TDM double-buffered A (bf16), packed B, f32 accumulate ----------
// blockIdx.y = group g -> columns [g*80, g*80+80), B tiles (g*5+j), j=0..4
__global__ __launch_bounds__(256) void wmma_gemm_kernel(
    const unsigned short* __restrict__ Abf, int K, int M, int ktiles,
    const unsigned short* __restrict__ Bp,
    float* __restrict__ D, int ldd,
    const float* __restrict__ bias) {
    __shared__ __align__(16) unsigned short As[2][5120];  // 2 x (128 rows x 40 ushorts)
    const int tid = threadIdx.x;
    const int wave = tid >> 5, lane = tid & 31;
    const int m0 = blockIdx.x * 128;
    const int g  = blockIdx.y;
    const int n    = lane & 15;
    const int half = lane >> 4;
    const unsigned rows_valid = (unsigned)(M - m0);   // grid guarantees m0 < M
    const unsigned short* Arow0 = Abf + (size_t)m0 * K;

    if (wave == 0)
        tdm_load_tile((unsigned)(uintptr_t)&As[0][0], Arow0, rows_valid, (unsigned)K);

    v8f acc[5] = {};
    for (int kt = 0; kt < ktiles; ++kt) {
        if (wave == 0) __builtin_amdgcn_s_wait_tensorcnt(0);   // tile kt landed in LDS
        __syncthreads();                                       // publish to all waves
        if (wave == 0 && kt + 1 < ktiles)                      // prefetch kt+1 into other buf
            tdm_load_tile((unsigned)(uintptr_t)&As[(kt + 1) & 1][0],
                          Arow0 + (size_t)(kt + 1) * 32, rows_valid, (unsigned)K);

        const unsigned short* asb = &As[kt & 1][0];
        const int arow = wave * 16 + n;
        union { uint4 q[2]; v16bf v; } afr;
        afr.q[0] = *(const uint4*)&asb[arow * 40 + half * 8];
        afr.q[1] = *(const uint4*)&asb[arow * 40 + 16 + half * 8];
        #pragma unroll
        for (int j = 0; j < 5; ++j) {
            const unsigned short* bp =
                Bp + ((size_t)((g * 5 + j) * ktiles + kt) * 32 + lane) * 16;
            union { uint4 q[2]; v16bf v; } bfr;
            bfr.q[0] = *(const uint4*)(bp);
            bfr.q[1] = *(const uint4*)(bp + 8);
            acc[j] = __builtin_amdgcn_wmma_f32_16x16x32_bf16(
                false, afr.v, false, bfr.v, (short)0, acc[j], false, false);
        }
        // reads of As complete before this wave reaches the next barrier,
        // so the next prefetch (other buffer) can never overtake a reader.
    }
    // store D (C/D layout: VGPR r -> M = r + 8*half, N = lane&15)
    #pragma unroll
    for (int j = 0; j < 5; ++j) {
        int col = g * 80 + j * 16 + n;
        float bv = 0.f;
        if (bias != nullptr && (j * 16 + n) < 78) bv = bias[j * 16 + n];
        #pragma unroll
        for (int r = 0; r < 8; ++r) {
            int row = m0 + wave * 16 + r + half * 8;
            if (row < M) D[(size_t)row * ldd + col] = acc[j][r] + bv;
        }
    }
}

// ---------- per-node attention dot products ----------
__global__ void dots_col_kernel(const float* __restrict__ Hc,
                                const float* __restrict__ attn_l, const float* __restrict__ attn_r,
                                float* __restrict__ er, float* __restrict__ el1,
                                float* __restrict__ el2, int N) {
    int wave = threadIdx.x >> 5, lane = threadIdx.x & 31;
    int node = blockIdx.x * 8 + wave;
    if (node >= N) return;
    const float* h = Hc + (size_t)node * 400;
    float e0 = 0, e1 = 0, e2 = 0, e4 = 0, l1 = 0, l2 = 0;
    for (int c = lane; c < 78; c += 32) {
        float h0 = h[c], h1v = h[80 + c], h2v = h[160 + c], h4v = h[320 + c];
        e0 += h0  * attn_r[c];
        e1 += h1v * attn_r[78 + c];
        e2 += h2v * attn_r[156 + c];
        e4 += h4v * attn_r[312 + c];
        l1 += h1v * attn_l[78 + c];
        l2 += h2v * attn_l[156 + c];
    }
    e0 = wred(e0); e1 = wred(e1); e2 = wred(e2); e4 = wred(e4); l1 = wred(l1); l2 = wred(l2);
    if (lane == 0) {
        er[node] = e0; er[N + node] = e1; er[2 * N + node] = e2; er[3 * N + node] = e4;
        el1[node] = l1; el2[node] = l2;
    }
}

__global__ void dot80_kernel(const float* __restrict__ S, const float* __restrict__ vec,
                             float* __restrict__ out, int N) {
    int wave = threadIdx.x >> 5, lane = threadIdx.x & 31;
    int node = blockIdx.x * 8 + wave;
    if (node >= N) return;
    float a = 0.f;
    for (int c = lane; c < 78; c += 32) a += S[(size_t)node * 80 + c] * vec[c];
    a = wred(a);
    if (lane == 0) out[node] = a;
}

// ---------- init ----------
__global__ void init_mz_kernel(unsigned* __restrict__ menc, float* __restrict__ z, int n) {
    int i = blockIdx.x * 256 + threadIdx.x;
    if (i < n) { menc[i] = 0x007FFFFFu; /* enc(-inf) */ z[i] = 0.f; }
}

__global__ void init_out_kernel(const float* __restrict__ Hc, const float* __restrict__ bgat,
                                float* __restrict__ out, int N) {
    int i = blockIdx.x * 256 + threadIdx.x;
    if (i >= N * 78) return;
    int n = i / 78, c = i % 78;
    float b = bgat[c] + bgat[78 + c] + bgat[156 + c] + bgat[234 + c] + bgat[312 + c];
    out[i] = Hc[(size_t)n * 400 + 240 + c] + b;   // 'self' etype slot 3 + summed biases
}

// ---------- edge softmax passes ----------
__global__ void edge_max_kernel(const float* __restrict__ el, const float* __restrict__ er,
                                const int* __restrict__ src, const int* __restrict__ dst,
                                int E, float* __restrict__ ebuf, unsigned* __restrict__ menc) {
    int i = blockIdx.x * 256 + threadIdx.x;
    if (i >= E) return;
    float e = el[src[i]] + er[dst[i]];
    e = (e > 0.f) ? e : 0.2f * e;                 // leaky relu, slope 0.2
    ebuf[i] = e;
    atomicMax(&menc[dst[i]], encf(e));
}

__global__ void edge_expsum_kernel(const int* __restrict__ dst, int E,
                                   float* __restrict__ ebuf, const unsigned* __restrict__ menc,
                                   float* __restrict__ z) {
    int i = blockIdx.x * 256 + threadIdx.x;
    if (i >= E) return;
    int d = dst[i];
    float ex = __expf(ebuf[i] - decf(menc[d]));
    ebuf[i] = ex;
    atomicAdd(&z[d], ex);
}

__global__ void rcp_kernel(float* __restrict__ z, int n) {
    int i = blockIdx.x * 256 + threadIdx.x;
    if (i < n) { float v = z[i]; z[i] = (v > 0.f) ? 1.f / v : 0.f; }
}

// one block (96 threads) per edge: out[dst] += fs[src] * alpha
__global__ void edge_scatter_kernel(const float* __restrict__ ebuf, const float* __restrict__ zr,
                                    const int* __restrict__ src, const int* __restrict__ dst,
                                    const float* __restrict__ fs, int ldfs,
                                    float* __restrict__ out) {
    int e = blockIdx.x;
    int c = threadIdx.x;
    if (c >= 78) return;
    int s = src[e], d = dst[e];
    float alpha = ebuf[e] * zr[d];
    atomicAdd(&out[(size_t)d * 78 + c], fs[(size_t)s * ldfs + c] * alpha);
}

// ---------- launcher ----------
extern "C" void kernel_launch(void* const* d_in, const int* in_sizes, int n_in,
                              void* d_out, int out_size, void* d_ws, size_t ws_size,
                              hipStream_t stream) {
    const float* col_feats   = (const float*)d_in[0];
    const float* table_feats = (const float*)d_in[1];
    const float* numfeat_raw = (const float*)d_in[2];
    const float* W_num       = (const float*)d_in[3];
    const float* b_num       = (const float*)d_in[4];
    const float* W_all       = (const float*)d_in[5];
    const float* attn_l      = (const float*)d_in[6];
    const float* attn_r      = (const float*)d_in[7];
    const float* b_gat       = (const float*)d_in[8];
    const int* tc_src  = (const int*)d_in[9];
    const int* tc_dst  = (const int*)d_in[10];
    const int* txt_src = (const int*)d_in[11];
    const int* txt_dst = (const int*)d_in[12];
    const int* nn_src  = (const int*)d_in[13];
    const int* nn_dst  = (const int*)d_in[14];
    const int* nf_src  = (const int*)d_in[15];
    const int* nf_dst  = (const int*)d_in[16];
    float* out = (float*)d_out;

    const int N_COL = in_sizes[0] / 768;
    const int N_TAB = in_sizes[1] / 768;
    const int N_NUM = in_sizes[2] / 192;
    const int E_TC = in_sizes[9], E_TXT = in_sizes[11], E_NN = in_sizes[13], E_NF = in_sizes[15];

    // ---- workspace carve ----
    uint8_t* ws = (uint8_t*)d_ws;
    size_t off = 0;
    auto take = [&](size_t bytes) -> void* {
        void* p = ws + off;
        off += (bytes + 255) & ~(size_t)255;
        return p;
    };
    float* Hc  = (float*)take((size_t)N_COL * 400 * 4);   // 5 etype slots, stride 80 each
    float* Ht  = (float*)take((size_t)N_TAB * 80 * 4);
    float* Fnf = (float*)take((size_t)N_NUM * 80 * 4);
    float* Wc  = (float*)take((size_t)192 * 78 * 4);
    float* bc  = (float*)take((size_t)78 * 4);
    unsigned short* BpA = (unsigned short*)take((size_t)25 * 24 * 512 * 2);
    unsigned short* BpC = (unsigned short*)take((size_t)5 * 6 * 512 * 2);
    unsigned short* Acol = (unsigned short*)take((size_t)N_COL * 768 * 2);  // bf16 activations
    unsigned short* Atab = (unsigned short*)take((size_t)N_TAB * 768 * 2);
    unsigned short* Anum = (unsigned short*)take((size_t)N_NUM * 192 * 2);
    float* er  = (float*)take((size_t)4 * N_COL * 4);      // er for etypes {0,1,2,4}
    float* el1 = (float*)take((size_t)N_COL * 4);
    float* el2 = (float*)take((size_t)N_COL * 4);
    float* el0 = (float*)take((size_t)N_TAB * 4);
    float* el4 = (float*)take((size_t)N_NUM * 4);
    unsigned* menc = (unsigned*)take((size_t)4 * N_COL * 4);
    float* z   = (float*)take((size_t)4 * N_COL * 4);
    float* ebuf = (float*)take((size_t)(E_TC + E_TXT + E_NN + E_NF) * 4);
    float* eb0 = ebuf;
    float* eb1 = eb0 + E_TC;
    float* eb2 = eb1 + E_TXT;
    float* eb4 = eb2 + E_NN;

    // ---- prep: W_comb, packed bf16 weights, bf16 activations ----
    wcomb_kernel<<<(192 * 78 + 78 + 255) / 256, 256, 0, stream>>>(
        W_num, b_num, W_all + (size_t)4 * 768 * 78, Wc, bc);
    pack_w_kernel<<<(5 * 5 * 24 * 32 + 255) / 256, 256, 0, stream>>>(W_all, BpA, 5, 24, 768, 78);
    pack_w_kernel<<<(1 * 5 * 6 * 32 + 255) / 256, 256, 0, stream>>>(Wc, BpC, 1, 6, 192, 78);
    {
        long long n8c = (long long)N_COL * 768 / 8;
        long long n8t = (long long)N_TAB * 768 / 8;
        long long n8n = (long long)N_NUM * 192 / 8;
        cvt_bf16_kernel<<<(unsigned)((n8c + 255) / 256), 256, 0, stream>>>(col_feats, Acol, n8c);
        cvt_bf16_kernel<<<(unsigned)((n8t + 255) / 256), 256, 0, stream>>>(table_feats, Atab, n8t);
        cvt_bf16_kernel<<<(unsigned)((n8n + 255) / 256), 256, 0, stream>>>(numfeat_raw, Anum, n8n);
    }

    // ---- fused projections (TDM-fed WMMA bf16, f32 accumulate) ----
    {
        dim3 gc((N_COL + 127) / 128, 5);
        wmma_gemm_kernel<<<gc, 256, 0, stream>>>(Acol, 768, N_COL, 24, BpA, Hc, 400, nullptr);
        dim3 gt((N_TAB + 127) / 128, 1);
        wmma_gemm_kernel<<<gt, 256, 0, stream>>>(Atab, 768, N_TAB, 24, BpA, Ht, 80, nullptr);
        dim3 gn((N_NUM + 127) / 128, 1);
        wmma_gemm_kernel<<<gn, 256, 0, stream>>>(Anum, 192, N_NUM, 6, BpC, Fnf, 80, bc);
    }

    // ---- attention dots ----
    dots_col_kernel<<<(N_COL + 7) / 8, 256, 0, stream>>>(Hc, attn_l, attn_r, er, el1, el2, N_COL);
    dot80_kernel<<<(N_TAB + 7) / 8, 256, 0, stream>>>(Ht, attn_l, el0, N_TAB);
    dot80_kernel<<<(N_NUM + 7) / 8, 256, 0, stream>>>(Fnf, attn_l + 4 * 78, el4, N_NUM);

    // ---- init segment state + seed output with self-etype + biases ----
    init_mz_kernel<<<(4 * N_COL + 255) / 256, 256, 0, stream>>>(menc, z, 4 * N_COL);
    init_out_kernel<<<((unsigned)(N_COL * 78) + 255) / 256, 256, 0, stream>>>(Hc, b_gat, out, N_COL);

    // ---- segment softmax: pass 1 (max) ----
    edge_max_kernel<<<(E_TC  + 255) / 256, 256, 0, stream>>>(el0, er,             tc_src,  tc_dst,  E_TC,  eb0, menc);
    edge_max_kernel<<<(E_TXT + 255) / 256, 256, 0, stream>>>(el1, er + N_COL,     txt_src, txt_dst, E_TXT, eb1, menc + N_COL);
    edge_max_kernel<<<(E_NN  + 255) / 256, 256, 0, stream>>>(el2, er + 2 * N_COL, nn_src,  nn_dst,  E_NN,  eb2, menc + 2 * N_COL);
    edge_max_kernel<<<(E_NF  + 255) / 256, 256, 0, stream>>>(el4, er + 3 * N_COL, nf_src,  nf_dst,  E_NF,  eb4, menc + 3 * N_COL);

    // ---- pass 2 (exp + sum) ----
    edge_expsum_kernel<<<(E_TC  + 255) / 256, 256, 0, stream>>>(tc_dst,  E_TC,  eb0, menc,             z);
    edge_expsum_kernel<<<(E_TXT + 255) / 256, 256, 0, stream>>>(txt_dst, E_TXT, eb1, menc + N_COL,     z + N_COL);
    edge_expsum_kernel<<<(E_NN  + 255) / 256, 256, 0, stream>>>(nn_dst,  E_NN,  eb2, menc + 2 * N_COL, z + 2 * N_COL);
    edge_expsum_kernel<<<(E_NF  + 255) / 256, 256, 0, stream>>>(nf_dst,  E_NF,  eb4, menc + 3 * N_COL, z + 3 * N_COL);
    rcp_kernel<<<(4 * N_COL + 255) / 256, 256, 0, stream>>>(z, 4 * N_COL);

    // ---- pass 3 (weighted scatter-add) ----
    edge_scatter_kernel<<<E_TC,  96, 0, stream>>>(eb0, z,             tc_src,  tc_dst,  Ht,       80,  out);
    edge_scatter_kernel<<<E_TXT, 96, 0, stream>>>(eb1, z + N_COL,     txt_src, txt_dst, Hc + 80,  400, out);
    edge_scatter_kernel<<<E_NN,  96, 0, stream>>>(eb2, z + 2 * N_COL, nn_src,  nn_dst,  Hc + 160, 400, out);
    edge_scatter_kernel<<<E_NF,  96, 0, stream>>>(eb4, z + 3 * N_COL, nf_src,  nf_dst,  Fnf,      80,  out);
}
